// ParallelTransformerBlock_1245540516041
// MI455X (gfx1250) — matrix operand order repaired and checked by reading, not compiled, not run on hardware
//
#include <hip/hip_runtime.h>
#include <hip/hip_bf16.h>

// ---------- types ----------
typedef __attribute__((ext_vector_type(16))) __bf16 bf16x16;
typedef __attribute__((ext_vector_type(8)))  float  f32x8;
typedef __attribute__((ext_vector_type(4)))  unsigned int u32x4;
typedef __attribute__((ext_vector_type(8)))  unsigned int u32x8;

static __device__ __forceinline__ unsigned short f2bf(float f) {
  union { float f; unsigned u; } x; x.f = f;
  unsigned u = x.u + 0x7FFFu + ((x.u >> 16) & 1u);
  return (unsigned short)(u >> 16);
}
static __device__ __forceinline__ float bf2f(unsigned short h) {
  union { unsigned u; float f; } x; x.u = ((unsigned)h) << 16;
  return x.f;
}
// Load a 16x(bf16) WMMA operand from two contiguous 16B chunks.
static __device__ __forceinline__ bf16x16 ldfrag(const unsigned short* p0,
                                                 const unsigned short* p1) {
  union { bf16x16 v; u32x4 q[2]; } u;
  u.q[0] = *(const u32x4*)p0;
  u.q[1] = *(const u32x4*)p1;
  return u.v;
}
static __device__ __forceinline__ f32x8 wmma_bf16(bf16x16 a, bf16x16 b, f32x8 c) {
  return __builtin_amdgcn_wmma_f32_16x16x32_bf16(false, a, false, b,
                                                 (short)0, c, false, false);
}
// LDS byte offset: low dword of the generic shared-aperture address (ISA 10.2).
static __device__ __forceinline__ unsigned lds_off_of(const void* p) {
  return (unsigned)(unsigned long long)p;
}
// Async global->LDS 128-bit copy (per-lane), tracked by ASYNCcnt.
static __device__ __forceinline__ void async_ld_b128(unsigned lds_off,
                                                     const void* gaddr) {
  asm volatile("global_load_async_to_lds_b128 %0, %1, off"
               :: "v"(lds_off), "v"((unsigned long long)gaddr)
               : "memory");
}
static __device__ __forceinline__ void wait_async0() {
  asm volatile("s_wait_asynccnt 0x0" ::: "memory");
}

// Tensor Data Mover: 2D tile (tile_w x tile_h elems, 2B elems) global -> LDS.
// D# layout per CDNA5 ISA 8.3 (group0) / 8.4 (group1); 2D => VADDR2/3 omitted.
static __device__ __forceinline__ void tdm_load_2d(unsigned lds_addr,
                                                   const void* gaddr,
                                                   unsigned tensor_w,
                                                   unsigned tensor_h,
                                                   unsigned tile_w,
                                                   unsigned tile_h,
                                                   unsigned row_stride_elems) {
  unsigned long long ga = (unsigned long long)gaddr;
  u32x4 g0;
  g0[0] = 1u;                                        // count=1 (valid user D#)
  g0[1] = lds_addr;                                  // lds_addr[31:0]
  g0[2] = (unsigned)ga;                              // global_addr[31:0]
  g0[3] = (unsigned)((ga >> 32) & 0x01FFFFFFu) | 0x80000000u; // addr[56:32]|type=2
  u32x8 g1;
  g1[0] = 1u << 16;                                  // data_size=1 (2 bytes)
  g1[1] = (tensor_w & 0xFFFFu) << 16;                // tensor_dim0[15:0]
  g1[2] = (tensor_w >> 16) | ((tensor_h & 0xFFFFu) << 16); // dim0 hi | dim1 lo
  g1[3] = (tensor_h >> 16) | (tile_w << 16);         // dim1 hi | tile_dim0
  g1[4] = tile_h & 0xFFFFu;                          // tile_dim1; tile_dim2=0
  g1[5] = row_stride_elems;                          // tensor_dim0_stride[31:0]
  g1[6] = 0u;                                        // stride hi | dim1_stride lo
  g1[7] = 0u;
  asm volatile("tensor_load_to_lds %0, %1" :: "s"(g0), "s"(g1) : "memory");
}

// ---------- constants ----------
constexpr int DIM      = 2048;
constexpr int HEADS    = 16;
constexpr int DH       = 128;
constexpr int FUSED    = 18688;
constexpr int FF       = 8192;
constexpr int MROWS    = 4096;   // B*N
constexpr int SEQ      = 2048;
constexpr int BM = 128, BN = 128, BK = 64;

// ---------- f32 -> bf16 convert ----------
__global__ __launch_bounds__(256) void k_cvt(const float* __restrict__ in,
                                             unsigned short* __restrict__ out,
                                             long n) {
  long i = (long)blockIdx.x * 256 + threadIdx.x;
  long stride = (long)gridDim.x * 256;
  for (; i < n; i += stride) out[i] = f2bf(in[i]);
}

// ---------- LayerNorm -> bf16 ----------
__global__ __launch_bounds__(256) void k_layernorm(const float* __restrict__ x,
                                                   const float* __restrict__ gamma,
                                                   const float* __restrict__ beta,
                                                   unsigned short* __restrict__ h) {
  __shared__ float sh[256], sh2[256];
  const int row = blockIdx.x, tid = threadIdx.x;
  const float* xr = x + (size_t)row * DIM;
  float s = 0.f, s2 = 0.f;
  for (int d = tid; d < DIM; d += 256) { float v = xr[d]; s += v; s2 += v * v; }
  sh[tid] = s; sh2[tid] = s2; __syncthreads();
  for (int off = 128; off > 0; off >>= 1) {
    if (tid < off) { sh[tid] += sh[tid + off]; sh2[tid] += sh2[tid + off]; }
    __syncthreads();
  }
  const float mu  = sh[0] * (1.0f / DIM);
  const float var = sh2[0] * (1.0f / DIM) - mu * mu;
  const float inv = rsqrtf(var + 1e-5f);
  for (int d = tid; d < DIM; d += 256)
    h[(size_t)row * DIM + d] = f2bf((xr[d] - mu) * inv * gamma[d] + beta[d]);
}

// ---------- TDM-fed double-buffered WMMA GEMM, bf16 out (h @ w_fused) ----------
// 128 threads = 4 waves, each wave computes a 64x64 sub-tile (4x4 WMMA accs).
// Wave 0 drives the Tensor Data Mover; TENSORcnt + barrier publish the tiles.
__global__ __launch_bounds__(128) void k_gemm1(const unsigned short* __restrict__ A,
                                               const unsigned short* __restrict__ Bm,
                                               unsigned short* __restrict__ C,
                                               int M, int N, int K) {
  __shared__ unsigned short As[2][BM][BK];   // 2 x 16KB
  __shared__ unsigned short Bs[2][BK][BN];   // 2 x 16KB
  const int tid  = threadIdx.x;
  const int lane = tid & 31, wave = tid >> 5;
  const int wm = (wave & 1) * 64;
  const int wn = (wave >> 1) * 64;
  const int l16 = lane & 15, hv = lane >> 4;
  const int mBase = blockIdx.y * BM;
  const int nBase = blockIdx.x * BN;

  f32x8 acc[4][4];
  const f32x8 z = {0.f,0.f,0.f,0.f,0.f,0.f,0.f,0.f};
  for (int i = 0; i < 4; ++i) for (int j = 0; j < 4; ++j) acc[i][j] = z;

  auto issue = [&](int kk, int buf) {   // wave 0 only: 2 TDM descriptors/tile
    tdm_load_2d(lds_off_of(&As[buf][0][0]),
                &A[(size_t)mBase * K + kk],
                (unsigned)K, (unsigned)M, BK, BM, (unsigned)K);
    tdm_load_2d(lds_off_of(&Bs[buf][0][0]),
                &Bm[(size_t)kk * N + nBase],
                (unsigned)N, (unsigned)K, BN, BK, (unsigned)N);
  };

  const int nT = K / BK;
  if (wave == 0) issue(0, 0);
  for (int t = 0; t < nT; ++t) {
    const int buf = t & 1;
    if (wave == 0) __builtin_amdgcn_s_wait_tensorcnt(0);
    __syncthreads();        // tiles in buf visible; all reads of buf^1 done
    if (wave == 0 && t + 1 < nT) issue((t + 1) * BK, buf ^ 1);
    #pragma unroll
    for (int ks = 0; ks < BK; ks += 32) {
      bf16x16 bfr[4];
      #pragma unroll
      for (int nt = 0; nt < 4; ++nt)
        for (int s = 0; s < 16; ++s)
          bfr[nt][s] = __builtin_bit_cast(__bf16, Bs[buf][ks + (s & 7) + 8 * hv + 16 * (s >> 3)][wn + nt * 16 + l16]);
      #pragma unroll
      for (int mt = 0; mt < 4; ++mt) {
        const int m = wm + mt * 16 + l16;
        bf16x16 a = ldfrag(&As[buf][m][ks + hv * 8],
                           &As[buf][m][ks + 16 + hv * 8]);
        #pragma unroll
        for (int nt = 0; nt < 4; ++nt)
          acc[mt][nt] = wmma_bf16(a, bfr[nt], acc[mt][nt]);
      }
    }
  }
  for (int mt = 0; mt < 4; ++mt)
    for (int nt = 0; nt < 4; ++nt)
      for (int r = 0; r < 8; ++r) {
        int row = mBase + wm + mt * 16 + r + 8 * hv;
        int col = nBase + wn + nt * 16 + l16;
        C[(size_t)row * N + col] = f2bf(acc[mt][nt][r]);
      }
}

// ---------- RoPE (q, kT) + copy v + SwiGLU ----------
__global__ __launch_bounds__(256) void k_rope_swiglu(const unsigned short* __restrict__ proj,
                                                     unsigned short* __restrict__ qr,
                                                     unsigned short* __restrict__ kT,
                                                     unsigned short* __restrict__ vb,
                                                     unsigned short* __restrict__ ffa) {
  const int row = blockIdx.x;            // b*2048 + n
  const int b = row >> 11, n = row & 2047;
  const unsigned short* pr = proj + (size_t)row * FUSED;
  const float LN1E4 = 9.210340371976184f;
  for (int d = threadIdx.x; d < DIM; d += 256) {
    int hh = d >> 7, dd = d & 127, base = d - dd, fi = dd & 63;
    float freq = __expf(-(float)(2 * fi) * (1.0f / 128.0f) * LN1E4);
    float pos = (float)n * freq;
    float c = __cosf(pos), s = __sinf(pos);
    float t = bf2f(pr[d]);
    float o = (dd < 64) ? -bf2f(pr[base + dd + 64]) : bf2f(pr[base + dd - 64]);
    qr[(((size_t)b * HEADS + hh) * SEQ + n) * DH + dd] = f2bf(t * c + o * s);
  }
  for (int dd = threadIdx.x; dd < DH; dd += 256) {
    int fi = dd & 63;
    float freq = __expf(-(float)(2 * fi) * (1.0f / 128.0f) * LN1E4);
    float pos = (float)n * freq;
    float c = __cosf(pos), s = __sinf(pos);
    float t = bf2f(pr[DIM + dd]);
    float o = (dd < 64) ? -bf2f(pr[DIM + dd + 64]) : bf2f(pr[DIM + dd - 64]);
    kT[((size_t)b * DH + dd) * SEQ + n] = f2bf(t * c + o * s);
    vb[(size_t)row * DH + dd] = pr[DIM + DH + dd];
  }
  for (int i = threadIdx.x; i < FF; i += 256) {
    float xf = bf2f(pr[DIM + 2 * DH + i]);
    float g  = bf2f(pr[DIM + 2 * DH + FF + i]);
    float sig = 1.0f / (1.0f + __expf(-g));
    ffa[(size_t)row * FF + i] = f2bf(g * sig * xf);
  }
}

// ---------- causal multi-query flash attention (1 wave / 16 query rows) ----------
__global__ __launch_bounds__(32) void k_attn(const unsigned short* __restrict__ qr,
                                             const unsigned short* __restrict__ kT,
                                             const unsigned short* __restrict__ vb,
                                             unsigned short* __restrict__ out) {
  __shared__ unsigned short Plds[16][32];
  const int lane = threadIdx.x & 31;
  const int l16 = lane & 15, hv = lane >> 4;
  const int qb = blockIdx.x, hd = blockIdx.y, b = blockIdx.z;
  const int qbase = qb * 16;
  const float scale = 0.08838834764831845f;  // 1/sqrt(128)

  bf16x16 qf[4];
  {
    const unsigned short* p =
        qr + (((size_t)b * HEADS + hd) * SEQ + qbase + l16) * DH;
    #pragma unroll
    for (int c = 0; c < 4; ++c)
      qf[c] = ldfrag(p + c * 32 + hv * 8, p + c * 32 + 16 + hv * 8);
  }
  const f32x8 z = {0.f,0.f,0.f,0.f,0.f,0.f,0.f,0.f};
  f32x8 O[8]; float mrow[8], lrow[8];
  for (int t = 0; t < 8; ++t) O[t] = z;
  for (int r = 0; r < 8; ++r) { mrow[r] = -3.0e38f; lrow[r] = 0.f; }

  const unsigned short* kTb = kT + (size_t)b * DH * SEQ;
  const unsigned short* vbb = vb + (size_t)b * SEQ * DH;
  const int jend = qbase + 15;
  for (int jbase = 0; jbase <= jend; jbase += 32) {
    f32x8 s0 = z, s1 = z;
    #pragma unroll
    for (int c = 0; c < 4; ++c) {
      bf16x16 kb0, kb1;
      for (int s = 0; s < 16; ++s) {
        const unsigned short* kp = kTb + (size_t)(c * 32 + (s & 7) + 8 * hv + 16 * (s >> 3)) * SEQ + jbase + l16;
        kb0[s] = __builtin_bit_cast(__bf16, kp[0]);
        kb1[s] = __builtin_bit_cast(__bf16, kp[16]);
      }
      s0 = wmma_bf16(qf[c], kb0, s0);
      s1 = wmma_bf16(qf[c], kb1, s1);
    }
    float corr[8];
    #pragma unroll
    for (int r = 0; r < 8; ++r) {
      const int rowg = qbase + r + 8 * hv;
      const int c0 = jbase + l16, c1 = jbase + 16 + l16;
      float a  = (c0 <= rowg) ? s0[r] * scale : -3.0e38f;
      float bb = (c1 <= rowg) ? s1[r] * scale : -3.0e38f;
      float rm = fmaxf(a, bb);
      for (int m = 1; m < 16; m <<= 1) rm = fmaxf(rm, __shfl_xor(rm, m, 32));
      const float mnew = fmaxf(mrow[r], rm);
      const float cr = __expf(mrow[r] - mnew);
      const float p0 = __expf(a - mnew);
      const float p1 = __expf(bb - mnew);
      float rs = p0 + p1;
      for (int m = 1; m < 16; m <<= 1) rs += __shfl_xor(rs, m, 32);
      lrow[r] = lrow[r] * cr + rs;
      mrow[r] = mnew; corr[r] = cr;
      const int prow = r + 8 * hv;
      Plds[prow][l16]      = f2bf(p0);
      Plds[prow][16 + l16] = f2bf(p1);
    }
    for (int t = 0; t < 8; ++t)
      for (int r = 0; r < 8; ++r) O[t][r] *= corr[r];
    __syncthreads();
    bf16x16 pf = ldfrag(&Plds[l16][hv * 8], &Plds[l16][16 + hv * 8]);
    const unsigned short* vp = vbb + (size_t)(jbase + 8 * hv) * DH + l16;
    #pragma unroll
    for (int t = 0; t < 8; ++t) {
      bf16x16 vf;
      for (int s = 0; s < 16; ++s)
        vf[s] = __builtin_bit_cast(__bf16, vp[(size_t)((s & 7) + 16 * (s >> 3)) * DH + t * 16]);
      O[t] = wmma_bf16(pf, vf, O[t]);
    }
    __syncthreads();
  }
  for (int t = 0; t < 8; ++t)
    for (int r = 0; r < 8; ++r) {
      const int rowg = qbase + r + 8 * hv;
      size_t idx = ((size_t)b * SEQ + rowg) * DIM + hd * DH + t * 16 + l16;
      out[idx] = f2bf(O[t][r] / lrow[r]);
    }
}

// ---------- dual-segment async double-buffered WMMA GEMM, f32 out ----------
__global__ __launch_bounds__(128) void k_gemm_out(const unsigned short* __restrict__ A0,
                                                  const unsigned short* __restrict__ B0, int K0,
                                                  const unsigned short* __restrict__ A1,
                                                  const unsigned short* __restrict__ B1, int K1,
                                                  float* __restrict__ C, int M, int N) {
  __shared__ unsigned short As[2][BM][BK];
  __shared__ unsigned short Bs[2][BK][BN];
  const int tid  = threadIdx.x;
  const int lane = tid & 31, wave = tid >> 5;
  const int wm = (wave & 1) * 64;
  const int wn = (wave >> 1) * 64;
  const int l16 = lane & 15, hv = lane >> 4;
  const int mBase = blockIdx.y * BM;
  const int nBase = blockIdx.x * BN;

  f32x8 acc[4][4];
  const f32x8 z = {0.f,0.f,0.f,0.f,0.f,0.f,0.f,0.f};
  for (int i = 0; i < 4; ++i) for (int j = 0; j < 4; ++j) acc[i][j] = z;

  for (int seg = 0; seg < 2; ++seg) {
    const unsigned short* A = seg ? A1 : A0;
    const unsigned short* Bm = seg ? B1 : B0;
    const int K = seg ? K1 : K0;

    auto issue = [&](int kk, int buf) {
      #pragma unroll
      for (int i = 0; i < 8; ++i) {
        int c = tid + i * 128;
        int r = c >> 3, col = (c & 7) * 8;
        async_ld_b128(lds_off_of(&As[buf][r][col]),
                      &A[(size_t)(mBase + r) * K + kk + col]);
      }
      #pragma unroll
      for (int i = 0; i < 8; ++i) {
        int c = tid + i * 128;
        int r = c >> 4, col = (c & 15) * 8;
        async_ld_b128(lds_off_of(&Bs[buf][r][col]),
                      &Bm[(size_t)(kk + r) * N + nBase + col]);
      }
    };

    const int nT = K / BK;
    issue(0, 0);
    for (int t = 0; t < nT; ++t) {
      const int buf = t & 1;
      wait_async0();
      __syncthreads();
      if (t + 1 < nT) issue((t + 1) * BK, buf ^ 1);
      #pragma unroll
      for (int ks = 0; ks < BK; ks += 32) {
        bf16x16 bfr[4];
        #pragma unroll
        for (int nt = 0; nt < 4; ++nt)
          for (int s = 0; s < 16; ++s)
            bfr[nt][s] = __builtin_bit_cast(__bf16, Bs[buf][ks + (s & 7) + 8 * hv + 16 * (s >> 3)][wn + nt * 16 + l16]);
        #pragma unroll
        for (int mt = 0; mt < 4; ++mt) {
          const int m = wm + mt * 16 + l16;
          bf16x16 a = ldfrag(&As[buf][m][ks + hv * 8],
                             &As[buf][m][ks + 16 + hv * 8]);
          #pragma unroll
          for (int nt = 0; nt < 4; ++nt)
            acc[mt][nt] = wmma_bf16(a, bfr[nt], acc[mt][nt]);
        }
      }
    }
  }
  for (int mt = 0; mt < 4; ++mt)
    for (int nt = 0; nt < 4; ++nt)
      for (int r = 0; r < 8; ++r) {
        int row = mBase + wm + mt * 16 + r + 8 * hv;
        int col = nBase + wn + nt * 16 + l16;
        C[(size_t)row * N + col] = acc[mt][nt][r];
      }
}

// ---------- host launcher ----------
extern "C" void kernel_launch(void* const* d_in, const int* in_sizes, int n_in,
                              void* d_out, int out_size, void* d_ws, size_t ws_size,
                              hipStream_t stream) {
  const float* x          = (const float*)d_in[0];
  const float* gamma      = (const float*)d_in[1];
  const float* beta       = (const float*)d_in[2];
  const float* w_fused    = (const float*)d_in[3];
  const float* w_attn_out = (const float*)d_in[4];
  const float* w_ff_out   = (const float*)d_in[5];
  float* out = (float*)d_out;
  (void)in_sizes; (void)n_in; (void)out_size; (void)ws_size;

  char* base = (char*)d_ws;
  size_t off = 0;
  auto carve = [&](size_t elems) -> unsigned short* {
    unsigned short* p = (unsigned short*)(base + off);
    off += (elems * 2 + 255) & ~(size_t)255;
    return p;
  };
  unsigned short* h_bf   = carve((size_t)MROWS * DIM);
  unsigned short* wf_bf  = carve((size_t)DIM * FUSED);
  unsigned short* wao_bf = carve((size_t)DIM * DIM);
  unsigned short* wfo_bf = carve((size_t)FF * DIM);
  unsigned short* proj   = carve((size_t)MROWS * FUSED);
  unsigned short* qr     = carve((size_t)2 * HEADS * SEQ * DH);
  unsigned short* kT     = carve((size_t)2 * DH * SEQ);
  unsigned short* vbuf   = carve((size_t)2 * SEQ * DH);
  unsigned short* ffa    = carve((size_t)MROWS * FF);
  unsigned short* attn   = carve((size_t)MROWS * DIM);

  k_cvt<<<2048, 256, 0, stream>>>(w_fused,    wf_bf,  (long)DIM * FUSED);
  k_cvt<<<1024, 256, 0, stream>>>(w_attn_out, wao_bf, (long)DIM * DIM);
  k_cvt<<<2048, 256, 0, stream>>>(w_ff_out,   wfo_bf, (long)FF * DIM);
  k_layernorm<<<MROWS, 256, 0, stream>>>(x, gamma, beta, h_bf);
  k_gemm1<<<dim3(FUSED / BN, MROWS / BM), 128, 0, stream>>>(h_bf, wf_bf, proj,
                                                            MROWS, FUSED, DIM);
  k_rope_swiglu<<<MROWS, 256, 0, stream>>>(proj, qr, kT, vbuf, ffa);
  k_attn<<<dim3(SEQ / 16, HEADS, 2), 32, 0, stream>>>(qr, kT, vbuf, attn);
  k_gemm_out<<<dim3(DIM / BN, MROWS / BM), 128, 0, stream>>>(
      attn, wao_bf, DIM, ffa, wfo_bf, FF, out, MROWS, DIM);
}
